// DeepGCN_80410377716005
// MI455X (gfx1250) — matrix operand order repaired and checked by reading, not compiled, not run on hardware
//
#include <hip/hip_runtime.h>
#include <hip/hip_bf16.h>

typedef __attribute__((ext_vector_type(16))) _Float16 v16h;
typedef __attribute__((ext_vector_type(8)))  float    v8f;
typedef unsigned int u32x4 __attribute__((ext_vector_type(4)));
typedef int          i32x4 __attribute__((ext_vector_type(4)));
typedef int          i32x8 __attribute__((ext_vector_type(8)));

#define NGN    23
#define HID    64
#define NLAY   20
#define EPER   256
#define NGRAPH 16384
#define ETOT   4194304
#define LN_EPS 1e-5f

// Pre-swizzled weight fragments: [layer][mat(2)][kblock(2)][ntile(4)][lane(32)] -> 16 f16 (32B)
#define NFRAG_BLK (NLAY * 16)          // 320 fragment tiles

// ---------------------------------------------------------------------------
// One-time weight swizzle into CDNA5 WMMA B-fragment order (f16).
// B 32x16 f16 fragment in this code's contiguous k order (A matches, not AMD's table order): lanes 0-15 -> N=lane, K=e; lanes 16-31 -> K=e+16.
// ---------------------------------------------------------------------------
__global__ __launch_bounds__(32)
void swizzle_weights(const float* __restrict__ W_rel,
                     const float* __restrict__ W_root,
                     v16h* __restrict__ wfrag) {
    const int bidx = blockIdx.x;            // l*16 + mat*8 + kb*4 + nt
    const int l   = bidx >> 4;
    const int mat = (bidx >> 3) & 1;
    const int kb  = (bidx >> 2) & 1;
    const int nt  = bidx & 3;
    const int lane = threadIdx.x;

    const float* W = (mat ? W_root : W_rel) + (size_t)l * HID * HID;
    const int n  = nt * 16 + (lane & 15);
    const int k0 = kb * 32 + ((lane >= 16) ? 16 : 0);
    v16h v;
#pragma unroll
    for (int e = 0; e < 16; ++e)
        v[e] = (_Float16)W[(k0 + e) * HID + n];
    wfrag[bidx * 32 + lane] = v;
}

// A 16x32 f16 fragment in this code's contiguous k order (same as B, not AMD's table order): lanes 0-15 -> M=lane, K=e;
// lanes 16-31 -> M=lane-16, K=e+16.
__device__ __forceinline__ v16h load_a_frag_f32(const float* __restrict__ src,
                                                int mtile, int kblock, int lane) {
    v16h a;
    const int m    = (lane & 15) + mtile * 16;
    const int koff = (lane >= 16) ? 16 : 0;
#pragma unroll
    for (int e = 0; e < 16; ++e) {
        const int k   = kblock * 32 + e + koff;
        a[e] = (_Float16)src[m * HID + k];
    }
    return a;
}

// ---------------------------------------------------------------------------
// One workgroup = one graph (23 nodes, 256 edges) through all 20 layers in LDS.
// ---------------------------------------------------------------------------
__global__ __launch_bounds__(256)
void gcn_graph_kernel(const float* __restrict__ x,
                      const int*   __restrict__ edge_index,
                      const float* __restrict__ W_enc,
                      const float* __restrict__ b_enc,
                      const float* __restrict__ ln_g,
                      const float* __restrict__ ln_b,
                      const float* __restrict__ b_rel,
                      const v16h*  __restrict__ wfrag,
                      float*       __restrict__ pooled) {
    __shared__ __align__(16) float hbuf[32 * HID];  // residual state fp32 (rows 23..31 pad)
    __shared__ __align__(16) float zbuf[32 * HID];  // relu(LN(h)) fp32
    __shared__ __align__(16) float aggb[32 * HID];  // segment-sum fp32
    __shared__ __align__(16) int   eraw[2 * EPER];  // TDM dest: row0 src gids, row1 dst gids
    __shared__ short ssrc[EPER];                    // edge srcs sorted by dst (CSR)
    __shared__ short edst[EPER];
    __shared__ int   rows[NGN + 1];                 // CSR row offsets
    __shared__ int   ofs[NGN];                      // scatter cursors
    __shared__ int   cnt[NGN + 1];

    const int g    = blockIdx.x;
    const int t    = threadIdx.x;
    const int lane = t & 31;
    const int wave = t >> 5;
    const int mtile = wave & 1;   // M rows [mtile*16, mtile*16+16)
    const int ntile = wave >> 1;  // N cols [ntile*16, ntile*16+16)

    // ---- TDM: DMA the 2x256 i32 edge tile (src row, dst row) into LDS ----
    // D# per CDNA5 ISA: count=1, type=2, data_size=4B,
    // tile 256x2, tensor_dim0_stride = E (row pitch between src[] and dst[]).
    if (wave == 0) {
        const unsigned long long ga =
            (unsigned long long)(uintptr_t)(edge_index + (size_t)g * EPER);
        const unsigned int lds_off = (unsigned int)(uintptr_t)eraw; // low 32b = LDS offset
        u32x4 g0;
        g0[0] = 1u;                                            // count=1 (valid), user mode
        g0[1] = lds_off;                                       // lds_addr
        g0[2] = (unsigned int)(ga & 0xFFFFFFFFu);              // global_addr[31:0]
        g0[3] = (unsigned int)((ga >> 32) & 0x1FFFFFFu)        // global_addr[56:32]
              | (2u << 30);                                    // type=2 ("image")
        i32x8 g1;
        g1[0] = (2 << 16);          // workgroup_mask=0, data_size=2 (4 bytes)
        g1[1] = (EPER << 16);       // tensor_dim0[15:0]=256 in bits[31:16]
        g1[2] = (2 << 16);          // tensor_dim0[31:16]=0, tensor_dim1=2
        g1[3] = (EPER << 16);       // tile_dim0=256 in bits[31:16]
        g1[4] = 2;                  // tile_dim1=2, tile_dim2=0
        g1[5] = (int)ETOT;          // tensor_dim0_stride[31:0] = E
        g1[6] = 0;                  // tensor_dim0_stride[47:32]=0
        g1[7] = 0;
        i32x4 gz = {0, 0, 0, 0};    // groups 2/3 unused (<=2D tensor)
#if __clang_major__ >= 23
        i32x8 gz8 = {0, 0, 0, 0, 0, 0, 0, 0};
        __builtin_amdgcn_tensor_load_to_lds(g0, g1, gz, gz, gz8, 0);
#else
        __builtin_amdgcn_tensor_load_to_lds(g0, g1, gz, gz, 0);
#endif
    }

    // ---- node encoder: h = x @ W_enc + b_enc (rows >= 23 zeroed) ----
#pragma unroll
    for (int it = 0; it < 8; ++it) {
        const int idx = it * 256 + t;
        const int r = idx >> 6, c = idx & 63;
        float acc = 0.0f;
        if (r < NGN) {
            const float* xr = x + (size_t)(g * NGN + r) * 8;
            acc = b_enc[c];
#pragma unroll
            for (int i = 0; i < 8; ++i) acc += xr[i] * W_enc[i * HID + c];
        }
        hbuf[idx] = acc;
    }
    // pad rows of aggregation buffer stay zero for the whole kernel
    for (int i = NGN * HID + t; i < 32 * HID; i += 256) aggb[i] = 0.0f;
    if (t < NGN + 1) cnt[t] = 0;

    if (wave == 0) __builtin_amdgcn_s_wait_tensorcnt(0);
    __syncthreads();   // edge tile + cnt zeros visible

    // ---- build CSR by dst (counting sort, once per WG) ----
    const short sl = (short)(eraw[t]        - g * NGN);  // local src
    const short dl = (short)(eraw[EPER + t] - g * NGN);  // local dst
    edst[t] = dl;
    atomicAdd(&cnt[(int)dl], 1);
    __syncthreads();
    if (t == 0) {
        int run = 0;
#pragma unroll
        for (int d = 0; d < NGN; ++d) {
            rows[d] = run; ofs[d] = run; run += cnt[d];
        }
        rows[NGN] = run;
    }
    __syncthreads();
    {
        const int p = atomicAdd(&ofs[(int)dl], 1);
        ssrc[p] = sl;
    }
    __syncthreads();

    // ---- 20 DeepGCN res+ layers ----
#pragma unroll 1
    for (int l = 0; l < NLAY; ++l) {
        // prefetch next layer's weight fragments (L2-resident, global_prefetch_b8)
        if (l + 1 < NLAY)
            __builtin_prefetch(&wfrag[((l + 1) * 16 + ntile) * 32 + lane], 0, 1);

        // z = relu(LayerNorm(h)): wave handles 4 rows via 32-lane shuffle reduce
        {
            const float* lg = ln_g + l * HID;
            const float* lb = ln_b + l * HID;
#pragma unroll
            for (int rr = 0; rr < 4; ++rr) {
                const int r = wave * 4 + rr;
                const float v0 = hbuf[r * HID + lane];
                const float v1 = hbuf[r * HID + lane + 32];
                float s = v0 + v1;
#pragma unroll
                for (int m = 16; m >= 1; m >>= 1) s += __shfl_xor(s, m, 32);
                const float mu = s * (1.0f / 64.0f);
                const float d0 = v0 - mu, d1 = v1 - mu;
                float q = d0 * d0 + d1 * d1;
#pragma unroll
                for (int m = 16; m >= 1; m >>= 1) q += __shfl_xor(q, m, 32);
                const float inv = rsqrtf(q * (1.0f / 64.0f) + LN_EPS);
                float z0 = fmaxf(d0 * inv * lg[lane]      + lb[lane],      0.0f);
                float z1 = fmaxf(d1 * inv * lg[lane + 32] + lb[lane + 32], 0.0f);
                if (r >= NGN) { z0 = 0.0f; z1 = 0.0f; }   // keep pad rows clean
                zbuf[r * HID + lane]      = z0;
                zbuf[r * HID + lane + 32] = z1;
            }
        }
        __syncthreads();

        // agg[d] = sum_{e in CSR row d} z[src[e]] : conflict-free float4 sums
#pragma unroll 1
        for (int idx = t; idx < NGN * 16; idx += 256) {
            const int d  = idx >> 4;
            const int f4 = (idx & 15) * 4;
            float4 acc = make_float4(0.f, 0.f, 0.f, 0.f);
            const int e1 = rows[d + 1];
            for (int e = rows[d]; e < e1; ++e) {
                const float4 zv = *(const float4*)&zbuf[(int)ssrc[e] * HID + f4];
                acc.x += zv.x; acc.y += zv.y; acc.z += zv.z; acc.w += zv.w;
            }
            *(float4*)&aggb[d * HID + f4] = acc;
        }
        __syncthreads();

        // h += agg @ W_rel + b_rel + z @ W_root  (v_wmma_f32_16x16x32_f16)
        {
            const v16h* wl = wfrag + (size_t)l * 16 * 32;   // this layer's fragments
            v8f c = {};
#pragma unroll
            for (int kb = 0; kb < 2; ++kb) {                // mat 0: W_rel
                v16h a = load_a_frag_f32(aggb, mtile, kb, lane);
                v16h b = wl[(kb * 4 + ntile) * 32 + lane];
                c = __builtin_amdgcn_wmma_f32_16x16x32_f16(false, a, false, b,
                                                           (short)0, c, false, false);
            }
#pragma unroll
            for (int kb = 0; kb < 2; ++kb) {                // mat 1: W_root
                v16h a = load_a_frag_f32(zbuf, mtile, kb, lane);
                v16h b = wl[(8 + kb * 4 + ntile) * 32 + lane];
                c = __builtin_amdgcn_wmma_f32_16x16x32_f16(false, a, false, b,
                                                           (short)0, c, false, false);
            }
            const int n = ntile * 16 + (lane & 15);
            const float brl = b_rel[l * HID + n];
            const int mbase = mtile * 16 + ((lane >= 16) ? 8 : 0);
#pragma unroll
            for (int r = 0; r < 8; ++r)
                hbuf[(mbase + r) * HID + n] += c[r] + brl;
        }
        __syncthreads();
    }

    // ---- per-graph mean pool over 23 real rows ----
    if (t < HID) {
        float s = 0.0f;
#pragma unroll
        for (int r = 0; r < NGN; ++r) s += hbuf[r * HID + t];
        pooled[(size_t)g * HID + t] = s * (1.0f / (float)NGN);
    }
}

// Temporal mean + 2-layer MLP head: one block per clip.
__global__ __launch_bounds__(64)
void head_kernel(const float* __restrict__ pooled,
                 const float* __restrict__ pos,
                 const float* __restrict__ W1, const float* __restrict__ b1,
                 const float* __restrict__ W2, const float* __restrict__ b2,
                 float* __restrict__ out) {
    __shared__ float v[HID];
    __shared__ float h1[HID];
    const int bclip = blockIdx.x;
    const int f = threadIdx.x;

    float s = 0.0f;
#pragma unroll
    for (int tt = 0; tt < 16; ++tt)
        s += pooled[(size_t)(bclip * 16 + tt) * HID + f] + pos[tt * HID + f];
    v[f] = s * (1.0f / 16.0f);
    __syncthreads();

    float a = b1[f];
#pragma unroll 8
    for (int k = 0; k < HID; ++k) a += v[k] * W1[k * HID + f];
    h1[f] = fmaxf(a, 0.0f);
    __syncthreads();

    if (f < 3) {
        float o = b2[f];
#pragma unroll 8
        for (int k = 0; k < HID; ++k) o += h1[k] * W2[k * 3 + f];
        out[bclip * 3 + f] = o;
    }
}

extern "C" void kernel_launch(void* const* d_in, const int* in_sizes, int n_in,
                              void* d_out, int out_size, void* d_ws, size_t ws_size,
                              hipStream_t stream) {
    const float* x     = (const float*)d_in[0];
    const int*   ei    = (const int*)  d_in[1];
    // d_in[2] = batch (unused: graph membership implicit, NGN nodes per graph)
    const float* Wenc  = (const float*)d_in[3];
    const float* benc  = (const float*)d_in[4];
    const float* lng   = (const float*)d_in[5];
    const float* lnb   = (const float*)d_in[6];
    const float* Wrel  = (const float*)d_in[7];
    const float* brel  = (const float*)d_in[8];
    const float* Wroot = (const float*)d_in[9];
    const float* pos   = (const float*)d_in[10];
    const float* W1    = (const float*)d_in[11];
    const float* b1    = (const float*)d_in[12];
    const float* W2    = (const float*)d_in[13];
    const float* b2    = (const float*)d_in[14];

    // workspace layout: [pooled 16384x64 f32 = 4 MiB][wfrag 320x32 v16h = 320 KiB]
    float* pooled = (float*)d_ws;
    v16h*  wfrag  = (v16h*)((char*)d_ws + (size_t)NGRAPH * HID * sizeof(float));
    float* out    = (float*)d_out;

    swizzle_weights<<<dim3(NFRAG_BLK), dim3(32), 0, stream>>>(Wrel, Wroot, wfrag);
    gcn_graph_kernel<<<dim3(NGRAPH), dim3(256), 0, stream>>>(
        x, ei, Wenc, benc, lng, lnb, brel, wfrag, pooled);
    head_kernel<<<dim3(1024), dim3(64), 0, stream>>>(
        pooled, pos, W1, b1, W2, b2, out);
}
